// RelationalNetwork_35175782154781
// MI455X (gfx1250) — matrix operand order repaired and checked by reading, not compiled, not run on hardware
//
#include <hip/hip_runtime.h>

typedef unsigned int   u32;
typedef unsigned short u16;
typedef __attribute__((ext_vector_type(16))) _Float16 v16h;
typedef __attribute__((ext_vector_type(8)))  float    v8f;

struct __attribute__((aligned(16))) U4 { u32 x, y, z, w; };
struct __attribute__((aligned(16))) U8 { U4 lo, hi; };

__device__ __forceinline__ u16 f2h(float f) {
  return __builtin_bit_cast(u16, (_Float16)f);
}
__device__ __forceinline__ u32 pk2(float a, float b) {
  return (u32)f2h(a) | ((u32)f2h(b) << 16);
}
__device__ __forceinline__ v16h mkfrag(U4 a, U4 b) {
  U8 t; t.lo = a; t.hi = b;
  return __builtin_bit_cast(v16h, t);
}

// ---------------------------------------------------------------------------
// Kernel 1: U[b,i,n] = xo[b,i,:66] @ W1[:66], V[b,i,n] = xo[b,i,:66] @ W1[66:132]
// xo = [x channels (64), ycoord, xcoord]. 524288 threads, one (b,i,n) each.
// ---------------------------------------------------------------------------
__global__ void __launch_bounds__(256) rn_prep_uv(
    const float* __restrict__ x, const float* __restrict__ w1,
    float* __restrict__ U, float* __restrict__ V) {
  int t = blockIdx.x * 256 + threadIdx.x;
  int n = t & 255;
  int i = (t >> 8) & 255;
  int b = t >> 16;
  const float* xr = x + (b * 64) * 256 + i;   // x[b,c,i], stride 256 over c
  float u = 0.f, v = 0.f;
#pragma unroll 4
  for (int c = 0; c < 64; ++c) {
    float f = xr[c * 256];
    u = fmaf(f, w1[c * 256 + n], u);
    v = fmaf(f, w1[(66 + c) * 256 + n], v);
  }
  float yy = -1.f + (float)(i >> 4) * (2.f / 15.f);
  float xx = -1.f + (float)(i & 15) * (2.f / 15.f);
  u = fmaf(yy, w1[64 * 256 + n], u);
  u = fmaf(xx, w1[65 * 256 + n], u);
  v = fmaf(yy, w1[130 * 256 + n], v);
  v = fmaf(xx, w1[131 * 256 + n], v);
  U[t] = u;
  V[t] = v;
}

// ---------------------------------------------------------------------------
// Kernel 2: Q[b,n] = code[b] @ W1[132:] + b1 ; pack W2/W3 fp32 -> f16 in WMMA
// B-fragment order; zero `summed`. B-frag order used here (contiguous k, same as A):
//   lane L: N = L%16, khalf = L/16 ; element e (0..15): K = kstep*32+16*khalf+e
// ---------------------------------------------------------------------------
__global__ void __launch_bounds__(256) rn_prep_qpack(
    const float* __restrict__ w1, const float* __restrict__ b1,
    const float* __restrict__ code, const float* __restrict__ w2,
    const float* __restrict__ w3, float* __restrict__ Qv,
    u16* __restrict__ W2p, u16* __restrict__ W3p, float* __restrict__ summed) {
  int t = blockIdx.x * 256 + threadIdx.x;
  if (t < 131072) {
    int p = t & 65535;
    int e = p & 15, lane = (p >> 4) & 31, kstep = (p >> 9) & 7, nt = p >> 12;
    int n = (nt << 4) | (lane & 15);
    int k = (kstep << 5) + ((lane >> 4) << 4) + e;
    const float* w = (t < 65536) ? w2 : w3;
    u16 hv = f2h(w[k * 256 + n]);
    if (t < 65536) W2p[p] = hv; else W3p[p] = hv;
  } else if (t < 133120) {
    int q = t - 131072;
    int b = q >> 8, n = q & 255;
    float acc = b1[n];
    for (int k = 0; k < 128; ++k)
      acc = fmaf(code[b * 128 + k], w1[(132 + k) * 256 + n], acc);
    Qv[q] = acc;
  } else if (t < 135168) {
    summed[t - 133120] = 0.f;
  }
}

// ---------------------------------------------------------------------------
// Kernel 3: fused g_theta layers 2+3. 2048 blocks x 256 thr (8 waves).
// Block = (b, i): all 256 j's for one i. Wave = 32 pairs (2 WMMA A-tiles),
// so each LDS B-fragment feeds 2 WMMAs. B fragments are double-buffered in
// registers so their ds_loads overlap the previous WMMA (no dscnt-0 stalls).
// Dynamic LDS: 128KB wave-private h tiles + 128KB staged weights + 1KB sums.
// ---------------------------------------------------------------------------
__global__ void __launch_bounds__(256) rn_main(
    const float* __restrict__ U, const float* __restrict__ V,
    const float* __restrict__ Qv, const u16* __restrict__ W2p,
    const u16* __restrict__ W3p, const float* __restrict__ b2,
    const float* __restrict__ b3, float* __restrict__ summed) {
  extern __shared__ char smem[];
  u16*   h_lds   = (u16*)smem;                        // 8*32*256 f16 = 128KB
  u16*   w_lds   = (u16*)(smem + 131072);             // 65536 f16 = 128KB
  float* sum_lds = (float*)(smem + 131072 + 131072);  // 256 f32

  const int tid  = threadIdx.x;
  const int lane = tid & 31;
  const int wave = tid >> 5;

  const int blk = blockIdx.x;
  const int b   = blk >> 8;
  const int i   = blk & 255;            // this block handles pair-rows (i, *)
  const int jbase = wave << 5;          // wave covers j in [jbase, jbase+32)
  const bool active = (jbase <= i);     // else entirely above the diagonal

  sum_lds[tid] = 0.f;

  // stage packed W2 into LDS (cooperative, 128KB)
  {
    const U4* src = (const U4*)W2p;
    U4* dst = (U4*)w_lds;
#pragma unroll
    for (int tt = 0; tt < 32; ++tt) dst[tid + tt * 256] = src[tid + tt * 256];
  }

  u16* hw = h_lds + wave * (32 * 256);
  const int c0 = lane << 3;  // 8 columns per lane

  if (active) {
    // h1[m, :] = relu(U[b,i,:] + V[b,jbase+m,:] + Q[b,:]) -> f16 rows in LDS
    const float4* Ur = (const float4*)(U + (((b << 8) + i) << 8));
    const float4* Qr = (const float4*)(Qv + (b << 8));
    float4 u0 = Ur[lane * 2], u1 = Ur[lane * 2 + 1];
    float4 q0 = Qr[lane * 2], q1 = Qr[lane * 2 + 1];
    for (int m = 0; m < 32; ++m) {
      const float4* Vr = (const float4*)(V + (((b << 8) + jbase + m) << 8));
      float4 v0 = Vr[lane * 2], v1 = Vr[lane * 2 + 1];
      float s0 = fmaxf(u0.x + v0.x + q0.x, 0.f);
      float s1 = fmaxf(u0.y + v0.y + q0.y, 0.f);
      float s2 = fmaxf(u0.z + v0.z + q0.z, 0.f);
      float s3 = fmaxf(u0.w + v0.w + q0.w, 0.f);
      float s4 = fmaxf(u1.x + v1.x + q1.x, 0.f);
      float s5 = fmaxf(u1.y + v1.y + q1.y, 0.f);
      float s6 = fmaxf(u1.z + v1.z + q1.z, 0.f);
      float s7 = fmaxf(u1.w + v1.w + q1.w, 0.f);
      U4 wv;
      wv.x = pk2(s0, s1); wv.y = pk2(s2, s3);
      wv.z = pk2(s4, s5); wv.w = pk2(s6, s7);
      *(U4*)(hw + m * 256 + c0) = wv;
    }
  }
  __syncthreads();  // W2 staged (h tiles are wave-private)

  const int Mrow  = lane & 15;
  const int khalf = lane >> 4;
  const int ncol  = lane & 15;
  const v8f vzero = {0.f, 0.f, 0.f, 0.f, 0.f, 0.f, 0.f, 0.f};
  v8f acc0[16], acc1[16];

  if (active) {
#pragma unroll
    for (int nt = 0; nt < 16; ++nt) { acc0[nt] = vzero; acc1[nt] = vzero; }
    for (int kstep = 0; kstep < 8; ++kstep) {
      // A frags (16x32 f16): lane holds M=lane%16, K runs k0..k0+15 (k0=kstep*32+16*khalf)
      const u16* ap0 = hw + Mrow * 256 + kstep * 32 + khalf * 16;
      v16h a0 = mkfrag(*(const U4*)ap0, *(const U4*)(ap0 + 8));
      const u16* ap1 = ap0 + 16 * 256;
      v16h a1 = mkfrag(*(const U4*)ap1, *(const U4*)(ap1 + 8));
      // B frags double-buffered: issue loads for nt+1 before WMMAs of nt
      const u16* wbase = w_lds + (kstep << 9) + (lane << 4);
      v16h bc = mkfrag(*(const U4*)wbase, *(const U4*)(wbase + 8));
#pragma unroll
      for (int nt = 0; nt < 16; ++nt) {
        v16h bn = bc;
        if (nt < 15) {
          const u16* bp = wbase + ((nt + 1) << 12);
          bn = mkfrag(*(const U4*)bp, *(const U4*)(bp + 8));
        }
        acc0[nt] = __builtin_amdgcn_wmma_f32_16x16x32_f16(
            false, a0, false, bc, (short)0, acc0[nt], false, false);
        acc1[nt] = __builtin_amdgcn_wmma_f32_16x16x32_f16(
            false, a1, false, bc, (short)0, acc1[nt], false, false);
        bc = bn;
      }
    }
    // h2 = relu(acc + b2) -> back into wave-private LDS rows (f16)
#pragma unroll
    for (int nt = 0; nt < 16; ++nt) {
      float bias = b2[(nt << 4) + ncol];
#pragma unroll
      for (int r = 0; r < 8; ++r) {
        int row = r + (khalf << 3);
        hw[row * 256 + (nt << 4) + ncol] = f2h(fmaxf(acc0[nt][r] + bias, 0.f));
        hw[(16 + row) * 256 + (nt << 4) + ncol] =
            f2h(fmaxf(acc1[nt][r] + bias, 0.f));
      }
    }
  }

  __syncthreads();  // everyone finished reading W2 from LDS
  {
    const U4* src = (const U4*)W3p;
    U4* dst = (U4*)w_lds;
#pragma unroll
    for (int tt = 0; tt < 32; ++tt) dst[tid + tt * 256] = src[tid + tt * 256];
  }
  __syncthreads();  // W3 staged, h2 visible

  if (active) {
#pragma unroll
    for (int nt = 0; nt < 16; ++nt) { acc0[nt] = vzero; acc1[nt] = vzero; }
    for (int kstep = 0; kstep < 8; ++kstep) {
      const u16* ap0 = hw + Mrow * 256 + kstep * 32 + khalf * 16;
      v16h a0 = mkfrag(*(const U4*)ap0, *(const U4*)(ap0 + 8));
      const u16* ap1 = ap0 + 16 * 256;
      v16h a1 = mkfrag(*(const U4*)ap1, *(const U4*)(ap1 + 8));
      const u16* wbase = w_lds + (kstep << 9) + (lane << 4);
      v16h bc = mkfrag(*(const U4*)wbase, *(const U4*)(wbase + 8));
#pragma unroll
      for (int nt = 0; nt < 16; ++nt) {
        v16h bn = bc;
        if (nt < 15) {
          const u16* bp = wbase + ((nt + 1) << 12);
          bn = mkfrag(*(const U4*)bp, *(const U4*)(bp + 8));
        }
        acc0[nt] = __builtin_amdgcn_wmma_f32_16x16x32_f16(
            false, a0, false, bc, (short)0, acc0[nt], false, false);
        acc1[nt] = __builtin_amdgcn_wmma_f32_16x16x32_f16(
            false, a1, false, bc, (short)0, acc1[nt], false, false);
        bc = bn;
      }
    }
    // rel = relu(acc + b3); keep rows with i >= j; partial-sum per column
#pragma unroll
    for (int nt = 0; nt < 16; ++nt) {
      float bias = b3[(nt << 4) + ncol];
      float part = 0.f;
#pragma unroll
      for (int r = 0; r < 8; ++r) {
        int j0 = jbase + r + (khalf << 3);
        float rv0 = fmaxf(acc0[nt][r] + bias, 0.f);
        float rv1 = fmaxf(acc1[nt][r] + bias, 0.f);
        part += (i >= j0) ? rv0 : 0.f;
        part += (i >= j0 + 16) ? rv1 : 0.f;
      }
      atomicAdd(&sum_lds[(nt << 4) + ncol], part);
    }
  }
  __syncthreads();
  unsafeAtomicAdd(&summed[(b << 8) + tid], sum_lds[tid]);
}

// ---------------------------------------------------------------------------
// Kernel 4: f_phi MLP on (8,256) in fp32 (negligible work). One block.
// ---------------------------------------------------------------------------
__global__ void __launch_bounds__(256) rn_fphi(
    const float* __restrict__ summed, const float* __restrict__ w1,
    const float* __restrict__ b1, const float* __restrict__ w2,
    const float* __restrict__ b2, const float* __restrict__ w3,
    const float* __restrict__ b3, float* __restrict__ out) {
  __shared__ float s0[256], s1[256], s2[256];
  const int tid = threadIdx.x;
  for (int b = 0; b < 8; ++b) {
    s0[tid] = summed[b * 256 + tid];
    __syncthreads();
    float a = b1[tid];
    for (int k = 0; k < 256; ++k) a = fmaf(s0[k], w1[k * 256 + tid], a);
    s1[tid] = fmaxf(a, 0.f);
    __syncthreads();
    a = b2[tid];
    for (int k = 0; k < 256; ++k) a = fmaf(s1[k], w2[k * 256 + tid], a);
    s2[tid] = fmaxf(a, 0.f);
    __syncthreads();
    if (tid < 32) {
      float o = b3[tid];
      for (int k = 0; k < 256; ++k) o = fmaf(s2[k], w3[k * 32 + tid], o);
      out[b * 32 + tid] = o;
    }
    __syncthreads();
  }
}

// ---------------------------------------------------------------------------
extern "C" void kernel_launch(void* const* d_in, const int* in_sizes, int n_in,
                              void* d_out, int out_size, void* d_ws,
                              size_t ws_size, hipStream_t stream) {
  (void)in_sizes; (void)n_in; (void)out_size; (void)ws_size;
  const float* x     = (const float*)d_in[0];
  const float* code  = (const float*)d_in[1];
  const float* gt_w1 = (const float*)d_in[2];
  const float* gt_b1 = (const float*)d_in[3];
  const float* gt_w2 = (const float*)d_in[4];
  const float* gt_b2 = (const float*)d_in[5];
  const float* gt_w3 = (const float*)d_in[6];
  const float* gt_b3 = (const float*)d_in[7];
  const float* fp_w1 = (const float*)d_in[8];
  const float* fp_b1 = (const float*)d_in[9];
  const float* fp_w2 = (const float*)d_in[10];
  const float* fp_b2 = (const float*)d_in[11];
  const float* fp_w3 = (const float*)d_in[12];
  const float* fp_b3 = (const float*)d_in[13];

  char*  ws     = (char*)d_ws;
  float* U      = (float*)ws;             // 524288 f32
  float* V      = U + 524288;             // 524288 f32
  float* Qv     = V + 524288;             // 2048 f32
  float* summed = Qv + 2048;              // 2048 f32
  u16*   W2p    = (u16*)(summed + 2048);  // 65536 f16
  u16*   W3p    = W2p + 65536;            // 65536 f16

  const int SMEM = 131072 + 131072 + 1024;  // 263168 B (<= 320KB WGP LDS)
  hipFuncSetAttribute((const void*)rn_main,
                      hipFuncAttributeMaxDynamicSharedMemorySize, SMEM);

  rn_prep_uv<<<2048, 256, 0, stream>>>(x, gt_w1, U, V);
  rn_prep_qpack<<<528, 256, 0, stream>>>(gt_w1, gt_b1, code, gt_w2, gt_w3, Qv,
                                         W2p, W3p, summed);
  rn_main<<<2048, 256, SMEM, stream>>>(U, V, Qv, W2p, W3p, gt_b2, gt_b3,
                                       summed);
  rn_fphi<<<1, 256, 0, stream>>>(summed, fp_w1, fp_b1, fp_w2, fp_b2, fp_w3,
                                 fp_b3, (float*)d_out);
}
